// DKernelPredefinedSparseAttention_66383014527457
// MI455X (gfx1250) — matrix operand order repaired and checked by reading, not compiled, not run on hardware
//
#include <hip/hip_runtime.h>

typedef __attribute__((ext_vector_type(16))) __bf16 v16bf;
typedef __attribute__((ext_vector_type(8)))  __bf16 v8bf;
typedef __attribute__((ext_vector_type(4)))  __bf16 v4bf;
typedef __attribute__((ext_vector_type(8)))  float  v8f;

#define H_    16
#define D_    128
#define S_    8192
#define BS_   64
#define NB_   (S_/BS_)
#define KMAX_ 8

// LDS strides in bf16 elements (padded for bank-conflict-free b128 access)
#define KSTR 136   // K block: 64 rows (key) x 136  (row stride 272B = 16*17)
#define VSTR 72    // V block transposed: 128 rows (d) x 72 (row stride 144B = 16*9)
#define PSTR 72    // per-wave P strip: 16 rows (m) x 72

union U16 { v16bf v; v8bf h[2]; };

__global__ __launch_bounds__(128)
void sparse_attn_fwd(const float* __restrict__ q, const float* __restrict__ k,
                     const float* __restrict__ v, const int* __restrict__ kidx,
                     float* __restrict__ out)
{
    __shared__ __bf16 sK[64 * KSTR];
    __shared__ __bf16 sV[D_ * VSTR];
    __shared__ __bf16 sP[4][16 * PSTR];

    const int qb   = blockIdx.x;
    const int h    = blockIdx.y;
    const int tid  = threadIdx.x;
    const int wv   = tid >> 5;       // wave id 0..3 -> 16-row query strip
    const int lane = tid & 31;
    const int hf   = lane >> 4;      // half of wave (0/1)
    const int l16  = lane & 15;

    // ---- Q strip -> 4 A-fragments (16x32 bf16 each), straight from global ----
    U16 aq[4];
    {
        const int  m    = wv * 16 + l16;                 // A layout: M = lane&15
        const long grow = (long)(qb * 64 + m) * H_ + h;
        const float* qrow = q + grow * D_;
        for (int kk = 0; kk < 4; ++kk) {
            // lanes 0-15: K = 0..7 & 16..23 ; lanes 16-31: K = 8..15 & 24..31
            const float* c1 = qrow + kk * 32 + 8 * hf;
            const float* c2 = c1 + 16;
            float4 f0 = *(const float4*)(c1);
            float4 f1 = *(const float4*)(c1 + 4);
            float4 f2 = *(const float4*)(c2);
            float4 f3 = *(const float4*)(c2 + 4);
            v16bf a;
            a[0]=(__bf16)f0.x; a[1]=(__bf16)f0.y; a[2]=(__bf16)f0.z; a[3]=(__bf16)f0.w;
            a[4]=(__bf16)f1.x; a[5]=(__bf16)f1.y; a[6]=(__bf16)f1.z; a[7]=(__bf16)f1.w;
            a[8]=(__bf16)f2.x; a[9]=(__bf16)f2.y; a[10]=(__bf16)f2.z; a[11]=(__bf16)f2.w;
            a[12]=(__bf16)f3.x; a[13]=(__bf16)f3.y; a[14]=(__bf16)f3.z; a[15]=(__bf16)f3.w;
            aq[kk].v = a;
        }
    }

    v8f co[8] = {};                      // O strip: 8 d-tiles, C layout (f32)
    float mrow[8], lrow[8];
    for (int r = 0; r < 8; ++r) { mrow[r] = -1e30f; lrow[r] = 0.0f; }

    const float scale = 0.08838834764831845f;    // 1/sqrt(128)
    const float L2E   = 1.4426950408889634f;

    int prev = -1;
    for (int t = 0; t < KMAX_; ++t) {
        const int kb = kidx[qb * KMAX_ + t];     // uniform across block
        if (t > 0 && kb <= prev) break;          // padding slots repeat -> stop
        prev = kb;

        __syncthreads();
        // ---- stage K (key-major) and V (transposed, d-major) as bf16 ----
        for (int p = 0; p < 16; ++p) {
            const int idx = p * 128 + tid;       // 2048 = 64 keys * 32 d4
            const int key = idx >> 5;
            const int d4  = idx & 31;
            const long gb = ((long)(kb * 64 + key) * H_ + h) * D_ + d4 * 4;
            const float4 kf = *(const float4*)(k + gb);
            const float4 vf = *(const float4*)(v + gb);
            v4bf kp;
            kp[0]=(__bf16)kf.x; kp[1]=(__bf16)kf.y; kp[2]=(__bf16)kf.z; kp[3]=(__bf16)kf.w;
            *(v4bf*)&sK[key * KSTR + d4 * 4] = kp;            // ds_store_b64
            sV[(d4 * 4 + 0) * VSTR + key] = (__bf16)vf.x;     // transposed scatter
            sV[(d4 * 4 + 1) * VSTR + key] = (__bf16)vf.y;
            sV[(d4 * 4 + 2) * VSTR + key] = (__bf16)vf.z;
            sV[(d4 * 4 + 3) * VSTR + key] = (__bf16)vf.w;
        }
        __syncthreads();

        // ---- S = Q * K^T : 4 N-tiles of 16x16, K-dim 128 = 4 wmma each ----
        v8f sc[4];
        for (int nt = 0; nt < 4; ++nt) {
            v8f c = {};
            for (int kk = 0; kk < 4; ++kk) {
                U16 b;   // B layout: N = lane&15 (key), K = 8*hf+0..7 & 16+8*hf+0..7
                const __bf16* bp = &sK[(nt * 16 + l16) * KSTR + kk * 32 + 8 * hf];
                b.h[0] = *(const v8bf*)(bp);
                b.h[1] = *(const v8bf*)(bp + 16);
                c = __builtin_amdgcn_wmma_f32_16x16x32_bf16(
                        false, aq[kk].v, false, b.v, (short)0, c, false, false);
            }
            sc[nt] = c;
        }

        for (int nt = 0; nt < 4; ++nt)
            for (int r = 0; r < 8; ++r) sc[nt][r] *= scale;

        if (kb == qb) {                           // causal mask, diagonal only
            for (int nt = 0; nt < 4; ++nt) {
                const int kpos = nt * 16 + l16;
                for (int r = 0; r < 8; ++r) {
                    const int qpos = wv * 16 + r + 8 * hf;
                    if (qpos < kpos) sc[nt][r] = -1e30f;
                }
            }
        }

        // ---- online softmax (rows spread over 16 lanes of each half) ----
        for (int r = 0; r < 8; ++r) {
            float mx = fmaxf(fmaxf(sc[0][r], sc[1][r]), fmaxf(sc[2][r], sc[3][r]));
            mx = fmaxf(mx, __shfl_xor(mx, 1, 16));
            mx = fmaxf(mx, __shfl_xor(mx, 2, 16));
            mx = fmaxf(mx, __shfl_xor(mx, 4, 16));
            mx = fmaxf(mx, __shfl_xor(mx, 8, 16));
            const float mnew = fmaxf(mrow[r], mx);
            const float a_   = exp2f((mrow[r] - mnew) * L2E);
            mrow[r] = mnew;
            float rs = 0.0f;
            for (int nt = 0; nt < 4; ++nt) {
                const float pv = exp2f((sc[nt][r] - mnew) * L2E);
                sc[nt][r] = pv;
                rs += pv;
            }
            rs += __shfl_xor(rs, 1, 16);
            rs += __shfl_xor(rs, 2, 16);
            rs += __shfl_xor(rs, 4, 16);
            rs += __shfl_xor(rs, 8, 16);
            lrow[r] = lrow[r] * a_ + rs;
            for (int dt = 0; dt < 8; ++dt) co[dt][r] *= a_;
        }

        // ---- C-layout P -> A-layout via per-wave LDS strip ----
        for (int nt = 0; nt < 4; ++nt)
            for (int r = 0; r < 8; ++r)
                sP[wv][(r + 8 * hf) * PSTR + nt * 16 + l16] = (__bf16)sc[nt][r];
        asm volatile("s_wait_dscnt 0" ::: "memory");

        U16 ap[2];
        for (int ks = 0; ks < 2; ++ks) {
            const __bf16* pp = &sP[wv][l16 * PSTR + ks * 32 + 8 * hf];
            ap[ks].h[0] = *(const v8bf*)(pp);
            ap[ks].h[1] = *(const v8bf*)(pp + 16);
        }

        // ---- O += P * V : 8 d-tiles, K-dim 64 = 2 wmma each ----
        for (int dt = 0; dt < 8; ++dt) {
            for (int ks = 0; ks < 2; ++ks) {
                U16 b;   // from transposed V: N = d (lane&15), K = key 8*hf+0..7 & 16+8*hf+0..7
                const __bf16* vp = &sV[(dt * 16 + l16) * VSTR + ks * 32 + 8 * hf];
                b.h[0] = *(const v8bf*)(vp);
                b.h[1] = *(const v8bf*)(vp + 16);
                co[dt] = __builtin_amdgcn_wmma_f32_16x16x32_bf16(
                            false, ap[ks].v, false, b.v, (short)0, co[dt], false, false);
            }
        }
    }

    // ---- normalize and write ----
    for (int r = 0; r < 8; ++r) {
        const float inv = 1.0f / lrow[r];
        const long row  = (long)(qb * 64 + wv * 16 + r + 8 * hf) * H_ + h;
        float* orow = out + row * D_;
        for (int dt = 0; dt < 8; ++dt)
            orow[dt * 16 + l16] = co[dt][r] * inv;
    }
}

extern "C" void kernel_launch(void* const* d_in, const int* in_sizes, int n_in,
                              void* d_out, int out_size, void* d_ws, size_t ws_size,
                              hipStream_t stream) {
    const float* q    = (const float*)d_in[0];
    const float* k    = (const float*)d_in[1];
    const float* v    = (const float*)d_in[2];
    const int*   kidx = (const int*)d_in[3];
    float* out = (float*)d_out;
    dim3 grid(NB_, H_);
    sparse_attn_fwd<<<grid, 128, 0, stream>>>(q, k, v, kidx, out);
}
